// FusionGNNNet_28003186770280
// MI455X (gfx1250) — matrix operand order repaired and checked by reading, not compiled, not run on hardware
//
#include <hip/hip_runtime.h>
#include <hip/hip_bf16.h>
#include <math.h>

typedef __attribute__((ext_vector_type(16))) _Float16 v16h;
typedef __attribute__((ext_vector_type(8)))  _Float16 v8h;
typedef __attribute__((ext_vector_type(8)))  float    v8f;
typedef __attribute__((ext_vector_type(4)))  unsigned int v4u;
typedef __attribute__((ext_vector_type(8)))  int      v8i;
typedef __attribute__((ext_vector_type(4)))  int      v4i;

#define BB 64
#define CC 64
#define TT 2000
#define TPAD 2048

__device__ __forceinline__ float geluf(float x) {
  return 0.5f * x * (1.f + erff(x * 0.70710678118654752f));
}
__device__ __forceinline__ float eluf(float x) {
  return x > 0.f ? x : expm1f(x);
}

template<int N>
__device__ float bsum(float v, float* red) {
  int tid = threadIdx.x;
  red[tid] = v; __syncthreads();
  #pragma unroll
  for (int s = N / 2; s > 0; s >>= 1) {
    if (tid < s) red[tid] += red[tid + s];
    __syncthreads();
  }
  float r = red[0]; __syncthreads();
  return r;
}
template<int N>
__device__ float bmax(float v, float* red) {
  int tid = threadIdx.x;
  red[tid] = v; __syncthreads();
  #pragma unroll
  for (int s = N / 2; s > 0; s >>= 1) {
    if (tid < s) red[tid] = fmaxf(red[tid], red[tid + s]);
    __syncthreads();
  }
  float r = red[0]; __syncthreads();
  return r;
}
template<int N>
__device__ float bmin(float v, float* red) {
  int tid = threadIdx.x;
  red[tid] = v; __syncthreads();
  #pragma unroll
  for (int s = N / 2; s > 0; s >>= 1) {
    if (tid < s) red[tid] = fminf(red[tid], red[tid + s]);
    __syncthreads();
  }
  float r = red[0]; __syncthreads();
  return r;
}

// ---------------------------------------------------------------------------
// Static adjacency: sym_norm(Adj + I)
// ---------------------------------------------------------------------------
__global__ void static_adj_kernel(const float* __restrict__ Adj, float* __restrict__ As) {
  __shared__ float rs[64];
  int tid = threadIdx.x; // 64 threads
  float s = 0.f;
  for (int d = 0; d < 64; ++d) s += Adj[tid * 64 + d] + (tid == d ? 1.f : 0.f);
  rs[tid] = s;
  __syncthreads();
  float ic = rsqrtf(fmaxf(rs[tid], 1e-6f));
  for (int d = 0; d < 64; ++d) {
    float v = Adj[tid * 64 + d] + (tid == d ? 1.f : 0.f);
    As[tid * 64 + d] = v * ic * rsqrtf(fmaxf(rs[d], 1e-6f));
  }
}

// ---------------------------------------------------------------------------
// Hand features + normalized-centered f16 rows (for WMMA gram matrix)
// ---------------------------------------------------------------------------
__global__ void handfeats_kernel(const float* __restrict__ x,
                                 float* __restrict__ feats,
                                 _Float16* __restrict__ xn) {
  __shared__ float row[TT];
  __shared__ float red[256];
  __shared__ float sh_m, sh_invn;
  int rid = blockIdx.x; // b*C + c
  const float* xr = x + (size_t)rid * TT;
  int tid = threadIdx.x;
  for (int t = tid; t < TT; t += 256) row[t] = xr[t];
  __syncthreads();

  float s1 = 0, s2 = 0, sa = 0, mx = -INFINITY, mn = INFINITY;
  for (int t = tid; t < TT; t += 256) {
    float v = row[t];
    s1 += v; s2 += v * v; sa += fabsf(v);
    mx = fmaxf(mx, v); mn = fminf(mn, v);
  }
  s1 = bsum<256>(s1, red);
  s2 = bsum<256>(s2, red);
  sa = bsum<256>(sa, red);
  mx = bmax<256>(mx, red);
  mn = bmin<256>(mn, red);

  float m  = s1 / (float)TT;
  float var = fmaxf(s2 / (float)TT - m * m, 0.f);
  float sd  = sqrtf(var);

  float s3 = 0, s4 = 0, zc = 0, sll = 0, sd1 = 0, sq1 = 0, sd2 = 0, sq2 = 0;
  for (int t = tid; t < TT; t += 256) {
    float v = row[t];
    float z = v - m;
    float z2 = z * z;
    s3 += z2 * z; s4 += z2 * z2;
    if (t < TT - 1) {
      float w = row[t + 1];
      float d1 = w - v;
      sll += fabsf(d1); sd1 += d1; sq1 += d1 * d1;
      float sg0 = (float)((v > 0.f) - (v < 0.f));
      float sg1 = (float)((w > 0.f) - (w < 0.f));
      zc += (sg0 != sg1) ? 1.f : 0.f;
      if (t < TT - 2) {
        float u = row[t + 2];
        float d2 = u - 2.f * w + v;
        sd2 += d2; sq2 += d2 * d2;
      }
    }
  }
  s3 = bsum<256>(s3, red); s4 = bsum<256>(s4, red);
  zc = bsum<256>(zc, red); sll = bsum<256>(sll, red);
  sd1 = bsum<256>(sd1, red); sq1 = bsum<256>(sq1, red);
  sd2 = bsum<256>(sd2, red); sq2 = bsum<256>(sq2, red);

  if (tid == 0) {
    float rms = sqrtf(s2 / (float)TT);
    float skew = (s3 / (float)TT) / (sd * sd * sd + 1e-6f);
    float kurt = (s4 / (float)TT) / (var * var + 1e-6f);
    float zcr = zc / (float)(TT - 1);
    float ll  = sll / (float)(TT - 1);
    float d1m = sd1 / (float)(TT - 1);
    float dsd = sqrtf(fmaxf(sq1 / (float)(TT - 1) - d1m * d1m, 0.f));
    float mob = dsd / (sd + 1e-6f);
    float d2m = sd2 / (float)(TT - 2);
    float d2sd = sqrtf(fmaxf(sq2 / (float)(TT - 2) - d2m * d2m, 0.f));
    float comp = d2sd / (dsd + 1e-6f) / (mob + 1e-6f);
    float* fo = feats + (size_t)rid * 16;
    fo[0] = m;  fo[1] = sd;  fo[2] = var; fo[3] = mx; fo[4] = mn;
    fo[5] = mx - mn; fo[6] = sa / (float)TT; fo[7] = rms;
    fo[8] = skew; fo[9] = kurt; fo[10] = zcr; fo[11] = ll;
    fo[12] = dsd; fo[13] = mob; fo[14] = comp; fo[15] = 0.f;
    sh_m = m;
    sh_invn = 1.f / (sqrtf(fmaxf(s2 - (float)TT * m * m, 0.f)) + 1e-6f);
  }
  __syncthreads();
  float mm = sh_m, invn = sh_invn;
  _Float16* xo = xn + (size_t)rid * TPAD;
  for (int t = tid; t < TPAD; t += 256)
    xo[t] = (_Float16)(t < TT ? (row[t] - mm) * invn : 0.f);
}

// ---------------------------------------------------------------------------
// Gram matrix sim[b,c,d] = |xn_c . xn_d| via v_wmma_f32_16x16x32_f16
// one wave per 16x16 tile; grid (16 tiles, B)
// ---------------------------------------------------------------------------
__global__ void gram_kernel(const _Float16* __restrict__ xn, float* __restrict__ sim) {
  int b = blockIdx.y;
  int tile = blockIdx.x;
  int c0 = (tile >> 2) * 16, d0 = (tile & 3) * 16;
  int lane = threadIdx.x;
  int r = lane & 15, hi = lane >> 4;
  const _Float16* xb = xn + (size_t)b * CC * TPAD;
  const _Float16* arow = xb + (size_t)(c0 + r) * TPAD;
  const _Float16* brow = xb + (size_t)(d0 + r) * TPAD;
  // pull the far halves of both rows toward the WGP early
  __builtin_prefetch(arow + 1024, 0, 0);
  __builtin_prefetch(brow + 1024, 0, 0);
  v8f acc = {};
  for (int k0 = 0; k0 < TPAD; k0 += 32) {
    union { v16h v; v8h h[2]; } A, Bm;
    A.h[0]  = *reinterpret_cast<const v8h*>(arow + k0 + hi * 8);
    A.h[1]  = *reinterpret_cast<const v8h*>(arow + k0 + 16 + hi * 8);
    Bm.h[0] = *reinterpret_cast<const v8h*>(brow + k0 + hi * 8);
    Bm.h[1] = *reinterpret_cast<const v8h*>(brow + k0 + 16 + hi * 8);
    acc = __builtin_amdgcn_wmma_f32_16x16x32_f16(false, A.v, false, Bm.v,
                                                 (short)0, acc, false, false);
  }
  float* so = sim + (size_t)b * CC * CC;
  #pragma unroll
  for (int i = 0; i < 8; ++i) {
    int c = c0 + i + hi * 8;
    so[c * CC + d0 + r] = fabsf(acc[i]);
  }
}

// ---------------------------------------------------------------------------
// Top-k mask per row
// ---------------------------------------------------------------------------
__global__ void topk_kernel(const float* __restrict__ sim, float* __restrict__ masked,
                            const int* __restrict__ topk_p) {
  int rid = blockIdx.x * blockDim.x + threadIdx.x;
  if (rid >= BB * CC) return;
  int K = topk_p[0];
  if (K < 1) K = 1; if (K > 64) K = 64;
  const float* sr = sim + (size_t)rid * CC;
  float v[64];
  for (int d = 0; d < 64; ++d) v[d] = sr[d];
  unsigned long long bits = 0ull;
  for (int it = 0; it < K; ++it) {
    float best = -INFINITY; int bi = 0;
    for (int d = 0; d < 64; ++d) { if (v[d] > best) { best = v[d]; bi = d; } }
    bits |= (1ull << bi);
    v[bi] = -INFINITY;
  }
  float* mo = masked + (size_t)rid * CC;
  for (int d = 0; d < 64; ++d) mo[d] = ((bits >> d) & 1ull) ? sr[d] : 0.f;
}

// ---------------------------------------------------------------------------
// Dynamic adjacency: A = 0.5(M+M^T)+I, sym_norm
// ---------------------------------------------------------------------------
__global__ void dynadj_kernel(const float* __restrict__ masked, float* __restrict__ adj) {
  __shared__ float Ab[4096];
  __shared__ float rs[64];
  int b = blockIdx.x, tid = threadIdx.x;
  if (tid < 64) rs[tid] = 0.f;
  __syncthreads();
  const float* mb = masked + (size_t)b * 4096;
  for (int idx = tid; idx < 4096; idx += 256) {
    int c = idx >> 6, d = idx & 63;
    float v = 0.5f * (mb[c * 64 + d] + mb[d * 64 + c]) + (c == d ? 1.f : 0.f);
    Ab[idx] = v;
    atomicAdd(&rs[c], v);
  }
  __syncthreads();
  float* ab = adj + (size_t)b * 4096;
  for (int idx = tid; idx < 4096; idx += 256) {
    int c = idx >> 6, d = idx & 63;
    ab[idx] = Ab[idx] * rsqrtf(fmaxf(rs[c], 1e-6f)) * rsqrtf(fmaxf(rs[d], 1e-6f));
  }
}

// ---------------------------------------------------------------------------
// Fully fused EEGNet per (b,c) row. Input row staged global->LDS by the
// Tensor Data Mover (TDM); temporal conv via WMMA implicit GEMM
// (M=16 padded filters x N=16 time positions x K=64 taps, 2 wmma per tile).
// ---------------------------------------------------------------------------
__global__ void eegnet_kernel(const float* __restrict__ x,
                              const float* __restrict__ tw,
                              const float* __restrict__ bn1g, const float* __restrict__ bn1b,
                              const float* __restrict__ pw,
                              const float* __restrict__ bn2g, const float* __restrict__ bn2b,
                              const float* __restrict__ dw,
                              const float* __restrict__ pw2,
                              const float* __restrict__ bn3g, const float* __restrict__ bn3b,
                              float* __restrict__ cnn) {
  __shared__ float    stage[TT];       // TDM destination (f32 row)
  __shared__ _Float16 xp[2080];        // padded input row (f16)
  __shared__ _Float16 twl[16 * 64];    // padded filter bank (rows 8..15 zero)
  __shared__ _Float16 y1[8 * 2016];    // temporal conv out (post bn1)
  __shared__ _Float16 h2[16 * 250];    // after pconv+bn2+elu+pool8
  __shared__ _Float16 dbuf[16 * 249];  // after depthwise conv
  __shared__ float accum[16];
  int rid = blockIdx.x; // b*C + c
  int tid = threadIdx.x;
  const float* xr = x + (size_t)rid * TT;

  // --- TDM async DMA of the 2000-float row into LDS (wave 0 only) ---------
  if (tid < 32) {
    unsigned long long ga = (unsigned long long)(uintptr_t)xr;   // 57-bit global addr
    unsigned int lds = (unsigned int)(uintptr_t)stage;           // addr[31:0] = LDS offset
    v4u g0;
    g0.x = 1u;                                                   // count=1 (valid user D#)
    g0.y = lds;                                                  // lds_addr
    g0.z = (unsigned int)ga;                                     // global_addr[31:0]
    g0.w = (unsigned int)((ga >> 32) & 0x01FFFFFFu) | (2u << 30); // addr[56:32], type=2
    v8i g1;
    g1[0] = 0x00020000;               // workgroup_mask=0, data_size=2 (4 bytes)
    g1[1] = (int)(2000u << 16);       // tensor_dim0[15:0] in bits[31:16]
    g1[2] = (int)(1u << 16);          // tensor_dim0[31:16]=0 | tensor_dim1 lo16 = 1
    g1[3] = (int)(2000u << 16);       // tensor_dim1 hi=0 | tile_dim0 = 2000
    g1[4] = 1;                        // tile_dim1=1 | tile_dim2=0
    g1[5] = 2000;                     // tensor_dim0_stride lo32
    g1[6] = (int)(2000u << 16);       // stride0 hi16=0 | tensor_dim1_stride lo16
    g1[7] = 0;                        // tensor_dim1_stride hi32
    v4i g2 = {0, 0, 0, 0};            // dims 2/3 unused (tile_dim3=0)
    v4i g3 = {0, 0, 0, 0};            // dim 4 unused
    v8i g4 = {0, 0, 0, 0, 0, 0, 0, 0}; // remaining descriptor words unused
    __builtin_amdgcn_tensor_load_to_lds(g0, g1, g2, g3, g4, 0);
    __builtin_amdgcn_s_wait_tensorcnt(0);
  }
  for (int i = tid; i < 16 * 64; i += 256)
    twl[i] = (_Float16)(i < 8 * 64 ? tw[i] : 0.f);
  if (tid < 16) accum[tid] = 0.f;
  __syncthreads();
  for (int i = tid; i < 2080; i += 256) {
    int s = i - 32;
    xp[i] = (_Float16)((s >= 0 && s < TT) ? stage[s] : 0.f);
  }
  __syncthreads();

  int lane = tid & 31, wave = tid >> 5;
  int r = lane & 15, hi = lane >> 4;
  const float bnscale = rsqrtf(1.001f); // bn2d eps = 1e-3
  for (int tile = wave; tile < 126; tile += 8) {
    int t0 = tile * 16;
    v8f acc = {};
    #pragma unroll
    for (int k0 = 0; k0 < 64; k0 += 32) {
      union { v16h v; v8h h[2]; } A, Bm;
      A.h[0] = *reinterpret_cast<const v8h*>(&twl[r * 64 + k0 + hi * 16]);
      A.h[1] = *reinterpret_cast<const v8h*>(&twl[r * 64 + k0 + hi * 16 + 8]);
      // B(k,n) = xp[t0 + n + k]; lane holds col n=r, K chunk by half-wave.
      int base = t0 + r + k0 + hi * 16;
      #pragma unroll
      for (int j = 0; j < 16; ++j) Bm.v[j] = xp[base + j];
      acc = __builtin_amdgcn_wmma_f32_16x16x32_f16(false, A.v, false, Bm.v,
                                                   (short)0, acc, false, false);
    }
    int t = t0 + r;
    #pragma unroll
    for (int i = 0; i < 8; ++i) {
      int f = i + hi * 8;
      if (f < 8 && t < 2001)
        y1[f * 2016 + t] = (_Float16)(acc[i] * (bn1g[f] * bnscale) + bn1b[f]);
    }
  }
  __syncthreads();

  // pointwise conv (8->16) + bn2 + elu + pool8 -> h2 (16 x 250)
  for (int idx = tid; idx < 16 * 250; idx += 256) {
    int g = idx / 250, p = idx % 250;
    float accp = 0.f;
    for (int j = 0; j < 8; ++j) {
      int t = p * 8 + j;
      float s = 0.f;
      #pragma unroll
      for (int f = 0; f < 8; ++f) s += pw[g * 8 + f] * (float)y1[f * 2016 + t];
      s = s * (bn2g[g] * bnscale) + bn2b[g];
      accp += eluf(s);
    }
    h2[g * 250 + p] = (_Float16)(accp * 0.125f);
  }
  __syncthreads();

  // depthwise temporal conv, 16 taps, pad 7 -> dbuf (16 x 249)
  for (int idx = tid; idx < 16 * 249; idx += 256) {
    int g = idx / 249, t = idx % 249;
    float s = 0.f;
    #pragma unroll
    for (int k = 0; k < 16; ++k) {
      int ti = t - 7 + k;
      if (ti >= 0 && ti < 250) s += dw[g * 16 + k] * (float)h2[g * 250 + ti];
    }
    dbuf[g * 249 + t] = (_Float16)s;
  }
  __syncthreads();

  // pw2 mix + bn3 + elu + pool8 + mean over windows == mean over t<248
  for (int idx = tid; idx < 16 * 248; idx += 256) {
    int o = idx / 248, t = idx % 248;
    float s = 0.f;
    #pragma unroll
    for (int g = 0; g < 16; ++g) s += pw2[o * 16 + g] * (float)dbuf[g * 249 + t];
    s = s * (bn3g[o] * bnscale) + bn3b[o];
    atomicAdd(&accum[o], eluf(s));
  }
  __syncthreads();
  if (tid < 16) cnn[(size_t)rid * 16 + tid] = accum[tid] / 248.f;
}

// ---------------------------------------------------------------------------
// Projection: nf(31) -> 128 (+LN+gelu) -> 64 (+LN)
// ---------------------------------------------------------------------------
__global__ void proj_kernel(const float* __restrict__ cnn, const float* __restrict__ feats,
                            const float* __restrict__ w1, const float* __restrict__ b1,
                            const float* __restrict__ g1, const float* __restrict__ bb1,
                            const float* __restrict__ w2, const float* __restrict__ b2,
                            const float* __restrict__ g2, const float* __restrict__ bb2,
                            float* __restrict__ hout) {
  __shared__ float nf[31];
  __shared__ float y[128];
  __shared__ float red[128];
  int rid = blockIdx.x, tid = threadIdx.x;
  if (tid < 16) nf[tid] = cnn[(size_t)rid * 16 + tid];
  else if (tid < 31) nf[tid] = feats[(size_t)rid * 16 + (tid - 16)];
  __syncthreads();
  float v = b1[tid];
  for (int i = 0; i < 31; ++i) v += w1[tid * 31 + i] * nf[i];
  float mean = bsum<128>(v, red) / 128.f;
  float z = v - mean;
  float var = bsum<128>(z * z, red) / 128.f;
  y[tid] = geluf(z * rsqrtf(var + 1e-5f) * g1[tid] + bb1[tid]);
  __syncthreads();
  float v2 = 0.f;
  if (tid < 64) {
    v2 = b2[tid];
    for (int i = 0; i < 128; ++i) v2 += w2[tid * 128 + i] * y[i];
  }
  float m2 = bsum<128>(tid < 64 ? v2 : 0.f, red) / 64.f;
  float z2 = v2 - m2;
  float var2 = bsum<128>(tid < 64 ? z2 * z2 : 0.f, red) / 64.f;
  if (tid < 64)
    hout[(size_t)rid * 64 + tid] = z2 * rsqrtf(var2 + 1e-5f) * g2[tid] + bb2[tid];
}

// ---------------------------------------------------------------------------
// GIN encoder (2 layers + JK), one workgroup per batch element
// ---------------------------------------------------------------------------
__global__ void gin_kernel(const float* __restrict__ hin,
                           const float* __restrict__ adj, int adjStride,
                           const float* __restrict__ eps,
                           const float* __restrict__ w1, const float* __restrict__ b1,
                           const float* __restrict__ bng, const float* __restrict__ bnb,
                           const float* __restrict__ w2, const float* __restrict__ b2,
                           const float* __restrict__ lng, const float* __restrict__ lnb,
                           const float* __restrict__ jw, const float* __restrict__ jb,
                           float* __restrict__ hout) {
  __shared__ float hb[4096];
  __shared__ float ob[4096];
  __shared__ _Float16 y1[8192];
  __shared__ _Float16 stk0[4096];
  int b = blockIdx.x, tid = threadIdx.x;
  const float* ad = adj + (size_t)b * adjStride;
  for (int i = tid; i < 4096; i += 256) hb[i] = hin[(size_t)b * 4096 + i];
  __syncthreads();
  const float bnscale = rsqrtf(1.0f + 1e-5f);
  for (int l = 0; l < 2; ++l) {
    float ep = 1.f + eps[l];
    const float* W1 = w1 + (size_t)l * 128 * 64;
    const float* B1 = b1 + l * 128;
    const float* G  = bng + l * 128;
    const float* Bb = bnb + l * 128;
    const float* W2 = w2 + (size_t)l * 64 * 128;
    const float* B2 = b2 + l * 64;
    const float* Lg = lng + l * 64;
    const float* Lb = lnb + l * 64;
    for (int idx = tid; idx < 4096; idx += 256) {
      int n = idx >> 6, f = idx & 63;
      float a = 0.f;
      for (int m = 0; m < 64; ++m) a += ad[n * 64 + m] * hb[m * 64 + f];
      ob[idx] = ep * hb[idx] + a;
    }
    __syncthreads();
    for (int idx = tid; idx < 8192; idx += 256) {
      int n = idx >> 7, j = idx & 127;
      float a = B1[j];
      for (int i = 0; i < 64; ++i) a += W1[j * 64 + i] * ob[n * 64 + i];
      a = a * (G[j] * bnscale) + Bb[j];
      y1[idx] = (_Float16)geluf(a);
    }
    __syncthreads();
    for (int idx = tid; idx < 4096; idx += 256) {
      int n = idx >> 6, f = idx & 63;
      float a = B2[f];
      for (int i = 0; i < 128; ++i) a += W2[f * 128 + i] * (float)y1[n * 128 + i];
      ob[idx] = a + hb[idx];
    }
    __syncthreads();
    if (tid < 64) {
      int n = tid;
      float m = 0.f;
      for (int f = 0; f < 64; ++f) m += ob[n * 64 + f];
      m /= 64.f;
      float var = 0.f;
      for (int f = 0; f < 64; ++f) { float zz = ob[n * 64 + f] - m; var += zz * zz; }
      var /= 64.f;
      float inv = rsqrtf(var + 1e-5f);
      for (int f = 0; f < 64; ++f) {
        float v = (ob[n * 64 + f] - m) * inv * Lg[f] + Lb[f];
        hb[n * 64 + f] = v;
        if (l == 0) stk0[n * 64 + f] = (_Float16)v;
      }
    }
    __syncthreads();
  }
  // Jumping-knowledge: sigmoid-gated sum of the two layer outputs
  if (tid < 64) {
    int n = tid;
    float d0 = jb[0], d1 = jb[0];
    for (int f = 0; f < 64; ++f) {
      d0 += (float)stk0[n * 64 + f] * jw[f];
      d1 += hb[n * 64 + f] * jw[f];
    }
    float w0 = 1.f / (1.f + expf(-d0));
    float w1s = 1.f / (1.f + expf(-d1));
    for (int f = 0; f < 64; ++f)
      hout[(size_t)b * 4096 + n * 64 + f] = w0 * (float)stk0[n * 64 + f] + w1s * hb[n * 64 + f];
  }
}

// ---------------------------------------------------------------------------
// Gating fusion + pooling + classifier head
// ---------------------------------------------------------------------------
__global__ void head_kernel(const float* __restrict__ hs_g, const float* __restrict__ hd_g,
                            const float* __restrict__ g1w, const float* __restrict__ g1b,
                            const float* __restrict__ g2w, const float* __restrict__ g2b,
                            const float* __restrict__ lng, const float* __restrict__ lnb,
                            const float* __restrict__ c1w, const float* __restrict__ c1b,
                            const float* __restrict__ c2w, const float* __restrict__ c2b,
                            float* __restrict__ out) {
  __shared__ float hs[4096], hd[4096], fu[4096];
  __shared__ float gr[128], grn[128], c1[64];
  __shared__ float lm, lvv;
  int b = blockIdx.x, tid = threadIdx.x;
  for (int i = tid; i < 4096; i += 256) {
    hs[i] = hs_g[(size_t)b * 4096 + i];
    hd[i] = hd_g[(size_t)b * 4096 + i];
  }
  __syncthreads();
  if (tid < 64) {
    int n = tid;
    float g1o[64];
    for (int j = 0; j < 64; ++j) {
      float a = g1b[j];
      for (int i = 0; i < 64; ++i) a += g1w[j * 128 + i] * hs[n * 64 + i];
      for (int i = 0; i < 64; ++i) a += g1w[j * 128 + 64 + i] * hd[n * 64 + i];
      g1o[j] = geluf(a);
    }
    float l0 = g2b[0], l1 = g2b[1];
    for (int j = 0; j < 64; ++j) { l0 += g2w[j] * g1o[j]; l1 += g2w[64 + j] * g1o[j]; }
    float mx = fmaxf(l0, l1);
    float e0 = expf(l0 - mx), e1 = expf(l1 - mx);
    float p0 = e0 / (e0 + e1), p1 = e1 / (e0 + e1);
    for (int f = 0; f < 64; ++f) fu[n * 64 + f] = p0 * hs[n * 64 + f] + p1 * hd[n * 64 + f];
  }
  __syncthreads();
  if (tid < 64) {
    int f = tid;
    float s = 0.f, mxv = -INFINITY;
    for (int n = 0; n < 64; ++n) { float v = fu[n * 64 + f]; s += v; mxv = fmaxf(mxv, v); }
    gr[f] = s / 64.f;
    gr[64 + f] = mxv;
  }
  __syncthreads();
  if (tid == 0) {
    float m = 0.f; for (int i = 0; i < 128; ++i) m += gr[i]; m /= 128.f;
    float v = 0.f; for (int i = 0; i < 128; ++i) { float z = gr[i] - m; v += z * z; }
    lm = m; lvv = v / 128.f;
  }
  __syncthreads();
  if (tid < 128) grn[tid] = (gr[tid] - lm) * rsqrtf(lvv + 1e-5f) * lng[tid] + lnb[tid];
  __syncthreads();
  if (tid < 64) {
    float a = c1b[tid];
    for (int i = 0; i < 128; ++i) a += c1w[tid * 128 + i] * grn[i];
    c1[tid] = geluf(a);
  }
  __syncthreads();
  if (tid < 4) {
    float a = c2b[tid];
    for (int i = 0; i < 64; ++i) a += c2w[tid * 64 + i] * c1[i];
    out[(size_t)b * 4 + tid] = a;
  }
}

// ---------------------------------------------------------------------------
extern "C" void kernel_launch(void* const* d_in, const int* in_sizes, int n_in,
                              void* d_out, int out_size, void* d_ws, size_t ws_size,
                              hipStream_t stream) {
  const float* x        = (const float*)d_in[0];
  const float* Adj      = (const float*)d_in[1];
  const float* tconv_w  = (const float*)d_in[2];
  const float* bn1_g    = (const float*)d_in[3];
  const float* bn1_b    = (const float*)d_in[4];
  const float* pconv_w  = (const float*)d_in[5];
  const float* bn2_g    = (const float*)d_in[6];
  const float* bn2_b    = (const float*)d_in[7];
  const float* dconv_w  = (const float*)d_in[8];
  const float* pw2_w    = (const float*)d_in[9];
  const float* bn3_g    = (const float*)d_in[10];
  const float* bn3_b    = (const float*)d_in[11];
  const float* proj1_w  = (const float*)d_in[12];
  const float* proj1_b  = (const float*)d_in[13];
  const float* ln1_g    = (const float*)d_in[14];
  const float* ln1_b    = (const float*)d_in[15];
  const float* proj2_w  = (const float*)d_in[16];
  const float* proj2_b  = (const float*)d_in[17];
  const float* ln2_g    = (const float*)d_in[18];
  const float* ln2_b    = (const float*)d_in[19];
  const float* gin_eps  = (const float*)d_in[20];
  const float* gin_w1   = (const float*)d_in[21];
  const float* gin_b1   = (const float*)d_in[22];
  const float* gin_bng  = (const float*)d_in[23];
  const float* gin_bnb  = (const float*)d_in[24];
  const float* gin_w2   = (const float*)d_in[25];
  const float* gin_b2   = (const float*)d_in[26];
  const float* gin_lng  = (const float*)d_in[27];
  const float* gin_lnb  = (const float*)d_in[28];
  const float* jk_w     = (const float*)d_in[29];
  const float* jk_b     = (const float*)d_in[30];
  const float* gate1_w  = (const float*)d_in[31];
  const float* gate1_b  = (const float*)d_in[32];
  const float* gate2_w  = (const float*)d_in[33];
  const float* gate2_b  = (const float*)d_in[34];
  const float* cls_ln_g = (const float*)d_in[35];
  const float* cls_ln_b = (const float*)d_in[36];
  const float* cls1_w   = (const float*)d_in[37];
  const float* cls1_b   = (const float*)d_in[38];
  const float* cls2_w   = (const float*)d_in[39];
  const float* cls2_b   = (const float*)d_in[40];
  const int*   topk     = (const int*)d_in[41];
  float* out = (float*)d_out;

  char* ws = (char*)d_ws;
  size_t off = 0;
  auto take = [&](size_t bytes) -> char* {
    char* p = ws + off;
    off += (bytes + 255) & ~(size_t)255;
    return p;
  };
  float*     feats  = (float*)take((size_t)BB * CC * 16 * sizeof(float));
  _Float16*  xn     = (_Float16*)take((size_t)BB * CC * TPAD * sizeof(_Float16));
  float*     sim    = (float*)take((size_t)BB * CC * CC * sizeof(float));
  float*     masked = (float*)take((size_t)BB * CC * CC * sizeof(float));
  float*     adjd   = (float*)take((size_t)BB * CC * CC * sizeof(float));
  float*     As     = (float*)take((size_t)CC * CC * sizeof(float));
  float*     cnn    = (float*)take((size_t)BB * CC * 16 * sizeof(float));
  float*     hbuf   = (float*)take((size_t)BB * CC * 64 * sizeof(float));
  float*     hsb    = (float*)take((size_t)BB * CC * 64 * sizeof(float));
  float*     hdb    = (float*)take((size_t)BB * CC * 64 * sizeof(float));

  static_adj_kernel<<<1, 64, 0, stream>>>(Adj, As);
  handfeats_kernel<<<BB * CC, 256, 0, stream>>>(x, feats, xn);
  gram_kernel<<<dim3(16, BB), 32, 0, stream>>>(xn, sim);
  topk_kernel<<<BB * CC / 64, 64, 0, stream>>>(sim, masked, topk);
  dynadj_kernel<<<BB, 256, 0, stream>>>(masked, adjd);
  eegnet_kernel<<<BB * CC, 256, 0, stream>>>(x, tconv_w, bn1_g, bn1_b, pconv_w,
                                             bn2_g, bn2_b, dconv_w, pw2_w, bn3_g, bn3_b, cnn);
  proj_kernel<<<BB * CC, 128, 0, stream>>>(cnn, feats, proj1_w, proj1_b, ln1_g, ln1_b,
                                           proj2_w, proj2_b, ln2_g, ln2_b, hbuf);
  for (int e = 0; e < 2; ++e) {
    gin_kernel<<<BB, 256, 0, stream>>>(
        hbuf,
        e == 0 ? As : adjd, e == 0 ? 0 : CC * CC,
        gin_eps + e * 2,
        gin_w1 + (size_t)e * 2 * 128 * 64, gin_b1 + e * 2 * 128,
        gin_bng + e * 2 * 128, gin_bnb + e * 2 * 128,
        gin_w2 + (size_t)e * 2 * 64 * 128, gin_b2 + e * 2 * 64,
        gin_lng + e * 2 * 64, gin_lnb + e * 2 * 64,
        jk_w + e * 64, jk_b + e,
        e == 0 ? hsb : hdb);
  }
  head_kernel<<<BB, 256, 0, stream>>>(hsb, hdb, gate1_w, gate1_b, gate2_w, gate2_b,
                                      cls_ln_g, cls_ln_b, cls1_w, cls1_b, cls2_w, cls2_b, out);
}
